// GCNAgg_36292473651327
// MI455X (gfx1250) — matrix operand order repaired and checked by reading, not compiled, not run on hardware
//
#include <hip/hip_runtime.h>

typedef __attribute__((ext_vector_type(16))) _Float16 v16h;
typedef __attribute__((ext_vector_type(8)))  float    v8f;

#define N_STACK 16
#define N_NODES 50000
#define D       256          // d_in == d_out == 256
#define NPB     64           // nodes per block

// Fused: y = mean_s(x) * 1/sqrt(17);  out = relu(y @ W^T + b)
// x: [16, 50000, 256] f32, W: [256, 256] f32 (row-major, i contiguous), b: [256] f32
__global__ __launch_bounds__(256) void gcn_fused_wmma(
    const float* __restrict__ x,
    const float* __restrict__ W,
    const float* __restrict__ bias,
    float* __restrict__ out) {

  __shared__ _Float16 sW[D * D];    // 128 KB : W in f16, row-major [o][i]
  __shared__ _Float16 sA[NPB * D];  //  32 KB : scaled mean in f16, [node][i]

  const int tid       = threadIdx.x;
  const int node_base = blockIdx.x * NPB;
  // mean over 16 stacks combined with 1/sqrt(17): 1/(16*sqrt(17))
  const float scale = 1.0f / (16.0f * 4.12310562561766f);

  // ---- Phase 1a: W fp32 -> f16 into LDS (each thread: 256 elems via float4) ----
  #pragma unroll 4
  for (int r = 0; r < 64; ++r) {
    const int idx = r * 1024 + tid * 4;           // covers 65536 elements
    float4 w = *(const float4*)(W + idx);
    union { _Float16 h[4]; uint2 u; } pk;
    pk.h[0] = (_Float16)w.x; pk.h[1] = (_Float16)w.y;
    pk.h[2] = (_Float16)w.z; pk.h[3] = (_Float16)w.w;
    *(uint2*)(&sW[idx]) = pk.u;
  }

  // ---- Phase 1b: stack mean (fp32 accumulate), scale, f16 -> LDS ----
  // thread -> (i0 = 4 consecutive features, nsub = node sub-index), 16 passes
  const int i0   = (tid & 63) * 4;
  const int nsub = tid >> 6;
  const size_t sstride = (size_t)N_NODES * D;     // stack stride in elements
  for (int t = 0; t < 16; ++t) {
    const int ln   = t * 4 + nsub;                // local node 0..63
    const int node = node_base + ln;
    float4 acc = make_float4(0.f, 0.f, 0.f, 0.f);
    if (node < N_NODES) {
      const float* p = x + ((size_t)node * D + i0);
      #pragma unroll
      for (int s = 0; s < N_STACK; ++s) {
        if (s + 2 < N_STACK) __builtin_prefetch(p + (size_t)(s + 2) * sstride, 0, 0);
        float4 v = *(const float4*)(p + (size_t)s * sstride);
        acc.x += v.x; acc.y += v.y; acc.z += v.z; acc.w += v.w;
      }
    }
    union { _Float16 h[4]; uint2 u; } pk;
    pk.h[0] = (_Float16)(acc.x * scale); pk.h[1] = (_Float16)(acc.y * scale);
    pk.h[2] = (_Float16)(acc.z * scale); pk.h[3] = (_Float16)(acc.w * scale);
    *(uint2*)(&sA[ln * D + i0]) = pk.u;
  }

  __syncthreads();

  // ---- Phase 2: WMMA GEMM. 8 waves: wave -> (node tile 0..3, output half 0..1) ----
  const int wave  = tid >> 5;
  const int lane  = tid & 31;
  const int ntile = wave >> 1;        // 16-node tile within block
  const int ohalf = wave & 1;         // which 8 of the 16 output tiles
  const int m     = lane & 15;        // row within A / col (N) within B,C,D
  const int kg    = lane >> 4;        // K-group select (ISA 16-bit fragment layout)

  // Load all 8 A fragments once (reused across the 8 output tiles).
  // A 16x32 f16: lanes 0-15 hold K = kb+0..7 (V0..3) and kb+16..23 (V4..7);
  //              lanes 16-31 hold K = kb+8..15 and kb+24..31.
  v16h Afrag[8];
  {
    const _Float16* arow = &sA[(ntile * 16 + m) * D];
    #pragma unroll
    for (int kt = 0; kt < 8; ++kt) {
      const int kb = kt * 32;
      union { uint4 u[2]; v16h h; } a;
      a.u[0] = *(const uint4*)(arow + kb + kg * 8);
      a.u[1] = *(const uint4*)(arow + kb + 16 + kg * 8);
      Afrag[kt] = a.h;
    }
  }

  #pragma unroll
  for (int ot8 = 0; ot8 < 8; ++ot8) {
    const int ot = ohalf * 8 + ot8;               // output tile (16 outputs each)
    const int o  = ot * 16 + m;                   // this lane's output column
    // B 32x16 f16: lane n=L%16, lanes 0-15 hold K=0..7,16..23, lanes 16-31 hold K=8..15,24..31,
    // matching the A fragment K order (two 8-half runs along i of W[o][i]).
    const _Float16* brow = &sW[o * D + kg * 8];

    v8f c = {};                                   // f32 accumulator
    #pragma unroll
    for (int kt = 0; kt < 8; ++kt) {
      union { uint4 u[2]; v16h h; } bfr;
      bfr.u[0] = *(const uint4*)(brow + kt * 32);
      bfr.u[1] = *(const uint4*)(brow + kt * 32 + 16);
      c = __builtin_amdgcn_wmma_f32_16x16x32_f16(
          /*neg_a=*/false, Afrag[kt], /*neg_b=*/false, bfr.h,
          /*c_mod=*/(short)0, c, /*reuse_a=*/false, /*reuse_b=*/false);
    }

    // D 16x16 f32: VGPR r -> M = r + 8*(lane/16), N = lane%16. Bias + ReLU + store.
    const float bv = bias[o];
    #pragma unroll
    for (int r = 0; r < 8; ++r) {
      const int node = node_base + ntile * 16 + r + kg * 8;
      if (node < N_NODES) {
        float v = c[r] + bv;
        out[(size_t)node * D + o] = v > 0.f ? v : 0.f;
      }
    }
  }
}

extern "C" void kernel_launch(void* const* d_in, const int* in_sizes, int n_in,
                              void* d_out, int out_size, void* d_ws, size_t ws_size,
                              hipStream_t stream) {
  const float* x = (const float*)d_in[0];
  const float* W = (const float*)d_in[1];
  const float* b = (const float*)d_in[2];
  float* out = (float*)d_out;
  (void)in_sizes; (void)n_in; (void)out_size; (void)d_ws; (void)ws_size;

  const int nblocks = (N_NODES + NPB - 1) / NPB;  // 782
  gcn_fused_wmma<<<dim3(nblocks), dim3(256), 0, stream>>>(x, W, b, out);
}
